// DCW_80547816669351
// MI455X (gfx1250) — matrix operand order repaired and checked by reading, not compiled, not run on hardware
//
#include <hip/hip_runtime.h>
#include <hip/hip_bf16.h>
#include <stdint.h>

typedef __attribute__((ext_vector_type(16))) __bf16 v16bf;
typedef __attribute__((ext_vector_type(8)))  __bf16 v8bf;
typedef __attribute__((ext_vector_type(8)))  float  v8f;
typedef __attribute__((ext_vector_type(4)))  float  v4f;
typedef __attribute__((ext_vector_type(4)))  unsigned int v4u;

static __device__ __forceinline__ v8bf bf8_zero() {
    v4u z = {0u, 0u, 0u, 0u};
    return __builtin_bit_cast(v8bf, z);
}
static __device__ __forceinline__ v8bf bf8_load(const __bf16* p) {
    v4u q = *reinterpret_cast<const v4u*>(p);
    return __builtin_bit_cast(v8bf, q);
}
// 8 f32 -> 8 bf16, non-temporal read (x is streamed exactly once)
static __device__ __forceinline__ v8bf bf8_cvt_nt(const float* p) {
    const v4f* q = reinterpret_cast<const v4f*>(p);
    v4f a = __builtin_nontemporal_load(q);
    v4f b = __builtin_nontemporal_load(q + 1);
    v8bf r;
    r[0] = (__bf16)a[0]; r[1] = (__bf16)a[1]; r[2] = (__bf16)a[2]; r[3] = (__bf16)a[3];
    r[4] = (__bf16)b[0]; r[5] = (__bf16)b[1]; r[6] = (__bf16)b[2]; r[7] = (__bf16)b[3];
    return r;
}
static __device__ __forceinline__ v16bf cat16(v8bf lo, v8bf hi) {
    return __builtin_shufflevector(lo, hi, 0,1,2,3,4,5,6,7,8,9,10,11,12,13,14,15);
}
static __device__ __forceinline__ v16bf bf16_load16(const __bf16* p) {
    v4u q0 = *reinterpret_cast<const v4u*>(p);
    v4u q1 = *reinterpret_cast<const v4u*>(p + 8);
    return cat16(__builtin_bit_cast(v8bf, q0), __builtin_bit_cast(v8bf, q1));
}

// ---------------------------------------------------------------------------
// Main fused GEMM: out[M,NPout] = act( in[M,K(stride KPin)] @ Wb[*,KPW]^T )
// Workgroup: 256 thr = 8 waves, 2(M) x 4(N); wave tile 32x32; block tile 64x128.
// Wb has >= blockIdx.y*128+128 rows (zero-padded) -> B loads unconditional.
// A staged via LDS (bf16); K handled as full 32-steps + one guarded tail.
// ---------------------------------------------------------------------------
template <bool IN_F32, bool OUT_F32, bool RELU>
__global__ __launch_bounds__(256) void mlp_gemm_kernel(
    const void* __restrict__ inp, const __bf16* __restrict__ Wb,
    void* __restrict__ outp, int M, int K, int KPin, int KPW, int NPout)
{
    __shared__ __bf16 As[64 * 32];

    const int tid  = threadIdx.x;
    const int lane = tid & 31;
    const int wave = tid >> 5;
    const int wm   = wave >> 2;   // 0..1
    const int wn   = wave & 3;    // 0..3
    const int lr   = lane & 15;   // row/col within 16
    const int lc   = lane >> 4;   // half-wave select

    const int mblk = blockIdx.x * 64;
    const int n0   = blockIdx.y * 128 + wn * 32;

    // staging: each thread moves 8 contiguous elements of one row (clamped)
    const int srow = tid >> 2;        // 0..63
    const int scol = (tid & 3) * 8;   // 0,8,16,24
    const int grow = (mblk + srow < M) ? (mblk + srow) : (M - 1);

    const float*  arow_f = (const float*)inp + (size_t)grow * KPin;
    const __bf16* arow_b = (const __bf16*)inp + (size_t)grow * KPin;
    __bf16* asdst = &As[srow * 32 + scol];
    const __bf16* asrc0 = &As[(wm * 32 + lr) * 32 + lc * 16];       // r=0
    const __bf16* asrc1 = &As[(wm * 32 + 16 + lr) * 32 + lc * 16];  // r=1
    const __bf16* brow0 = Wb + (size_t)(n0 + lr) * KPW + lc * 16;       // s=0
    const __bf16* brow1 = Wb + (size_t)(n0 + 16 + lr) * KPW + lc * 16;  // s=1

    v8f zf = {0.f,0.f,0.f,0.f,0.f,0.f,0.f,0.f};
    v8f acc[2][2];
    acc[0][0] = zf; acc[0][1] = zf; acc[1][0] = zf; acc[1][1] = zf;

    auto kstep = [&](int k0, bool guard) {
        __syncthreads();  // protect LDS vs previous iteration's fragment reads
        v8bf v;
        const int kk = k0 + scol;
        if (!guard || kk < K) {
            if constexpr (IN_F32) v = bf8_cvt_nt(arow_f + kk);
            else                  v = bf8_load(arow_b + kk);
        } else {
            v = bf8_zero();
        }
        *reinterpret_cast<v8bf*>(asdst) = v;
        __syncthreads();

        // A fragments: 16x32 bf16; lane = row M, 16 contiguous K per lane
        v16bf a0 = cat16(*reinterpret_cast<const v8bf*>(asrc0),
                         *reinterpret_cast<const v8bf*>(asrc0 + 8));
        v16bf a1 = cat16(*reinterpret_cast<const v8bf*>(asrc1),
                         *reinterpret_cast<const v8bf*>(asrc1 + 8));
        // B fragments: 32x16 bf16; lane = col N, 16 contiguous K per lane
        v16bf b0 = bf16_load16(brow0 + k0);
        v16bf b1 = bf16_load16(brow1 + k0);

        acc[0][0] = __builtin_amdgcn_wmma_f32_16x16x32_bf16(
            false, a0, false, b0, (short)0, acc[0][0], false, false);
        acc[0][1] = __builtin_amdgcn_wmma_f32_16x16x32_bf16(
            false, a0, false, b1, (short)0, acc[0][1], false, false);
        acc[1][0] = __builtin_amdgcn_wmma_f32_16x16x32_bf16(
            false, a1, false, b0, (short)0, acc[1][0], false, false);
        acc[1][1] = __builtin_amdgcn_wmma_f32_16x16x32_bf16(
            false, a1, false, b1, (short)0, acc[1][1], false, false);
    };

    const int kfull = K & ~31;
    for (int k0 = 0; k0 < kfull; k0 += 32) kstep(k0, false);
    if (K & 31) kstep(kfull, true);

    // C/D layout: VGPR v, lanes<16 -> M=v, lanes>=16 -> M=v+8; N=lane&15
#pragma unroll
    for (int r = 0; r < 2; ++r) {
        const int mbase = mblk + wm * 32 + r * 16 + lc * 8;
#pragma unroll
        for (int s = 0; s < 2; ++s) {
            const int n = n0 + s * 16 + lr;
            if (n < NPout) {
#pragma unroll
                for (int v = 0; v < 8; ++v) {
                    const int m = mbase + v;
                    if (m < M) {
                        float val = acc[r][s][v];
                        if (RELU) val = (val > 0.f) ? val : 0.01f * val;
                        const size_t idx = (size_t)m * NPout + n;
                        if constexpr (OUT_F32) {
                            __builtin_nontemporal_store(val, (float*)outp + idx);
                        } else {
                            ((__bf16*)outp)[idx] = (__bf16)val;
                        }
                    }
                }
            }
        }
    }
}

// ---------------------------------------------------------------------------
// Prep kernels (negligible cost): G = W W^T; Gauss-Jordan inverse; pinv rows.
// ---------------------------------------------------------------------------
__global__ void gram_kernel(const float* __restrict__ W, float* __restrict__ G,
                            int m, int k) {
    int idx = blockIdx.x * blockDim.x + threadIdx.x;
    if (idx >= m * m) return;
    int i = idx / m, j = idx % m;
    const float* ri = W + (size_t)i * k;
    const float* rj = W + (size_t)j * k;
    float acc = 0.f;
    for (int t = 0; t < k; ++t) acc = fmaf(ri[t], rj[t], acc);
    G[idx] = acc;
}

// Single-workgroup Gauss-Jordan on SPD G (m <= 128). LDS: [m][2m] + f[m].
__global__ void ginv_kernel(const float* __restrict__ G, float* __restrict__ Gi,
                            int m) {
    extern __shared__ float A[];
    float* f = A + m * 2 * m;
    const int tid  = threadIdx.x;
    const int twoM = 2 * m;
    for (int idx = tid; idx < m * twoM; idx += blockDim.x) {
        int i = idx / twoM, c = idx % twoM;
        A[idx] = (c < m) ? G[(size_t)i * m + c] : ((c - m) == i ? 1.f : 0.f);
    }
    __syncthreads();
    for (int p = 0; p < m; ++p) {
        const float invp = 1.f / A[(size_t)p * twoM + p];
        __syncthreads();
        for (int c = tid; c < twoM; c += blockDim.x) A[(size_t)p * twoM + c] *= invp;
        __syncthreads();
        for (int i = tid; i < m; i += blockDim.x) f[i] = A[(size_t)i * twoM + p];
        __syncthreads();
        for (int idx = tid; idx < m * twoM; idx += blockDim.x) {
            int i = idx / twoM, c = idx % twoM;
            if (i != p) A[idx] -= f[i] * A[(size_t)p * twoM + c];
        }
        __syncthreads();
    }
    for (int idx = tid; idx < m * m; idx += blockDim.x) {
        int i = idx / m, j = idx % m;
        Gi[idx] = A[(size_t)i * twoM + m + j];
    }
}

// out rows [0,nv): pinv(W)[a,b] = sum_j W[j,a]*Ginv[j,b] (b<m); rows/cols pad 0.
// out is [nAlloc, KP] bf16, launched over nAlloc*KP.
__global__ void pinv_bf_kernel(const float* __restrict__ W,
                               const float* __restrict__ Gi,
                               __bf16* __restrict__ out,
                               int m, int nv, int KP, int total) {
    int idx = blockIdx.x * blockDim.x + threadIdx.x;
    if (idx >= total) return;
    int a = idx / KP, b = idx % KP;
    float acc = 0.f;
    if (a < nv && b < m)
        for (int j = 0; j < m; ++j)
            acc = fmaf(W[(size_t)j * nv + a], Gi[(size_t)j * m + b], acc);
    out[idx] = (__bf16)acc;
}

// f32 [nv,k] -> bf16 [nAlloc,KP], zero-padded rows and cols
__global__ void cvt_bf_kernel(const float* __restrict__ W,
                              __bf16* __restrict__ out,
                              int nv, int k, int KP, int total) {
    int idx = blockIdx.x * blockDim.x + threadIdx.x;
    if (idx >= total) return;
    int a = idx / KP, b = idx % KP;
    out[idx] = (a < nv && b < k) ? (__bf16)W[(size_t)a * k + b] : (__bf16)0.f;
}

static inline int cdiv(int a, int b) { return (a + b - 1) / b; }

extern "C" void kernel_launch(void* const* d_in, const int* in_sizes, int n_in,
                              void* d_out, int out_size, void* d_ws, size_t ws_size,
                              hipStream_t stream) {
    const float* x  = (const float*)d_in[0];
    const float* W1 = (const float*)d_in[1];  // [128,784]
    const float* W2 = (const float*)d_in[2];  // [64,128]
    const float* W3 = (const float*)d_in[3];  // [16,64]
    const int B = in_sizes[0] / 784;

    // ---- scratch layout (256B aligned) ----
    size_t off = 0;
    auto take = [&](size_t bytes) -> void* {
        void* p = (char*)d_ws + off;
        off += (bytes + 255) & ~(size_t)255;
        return p;
    };
    float*  G1  = (float*)take(128 * 128 * sizeof(float));
    float*  G1i = (float*)take(128 * 128 * sizeof(float));
    float*  G2  = (float*)take(64 * 64 * sizeof(float));
    float*  G2i = (float*)take(64 * 64 * sizeof(float));
    float*  G3  = (float*)take(16 * 16 * sizeof(float));
    float*  G3i = (float*)take(16 * 16 * sizeof(float));
    // Weight buffers: rows padded to multiple of 128 (block N-tile), K padded to 32.
    __bf16* M1  = (__bf16*)take((size_t)128 * 800 * 2);  // W1       [128->128, 784->800]
    __bf16* M2  = (__bf16*)take((size_t)128 * 128 * 2);  // W2       [64->128, 128]
    __bf16* M3  = (__bf16*)take((size_t)128 * 64 * 2);   // W3       [16->128, 64]
    __bf16* M4  = (__bf16*)take((size_t)128 * 32 * 2);   // pinv(W3) [64->128, 16->32]
    __bf16* M5  = (__bf16*)take((size_t)128 * 64 * 2);   // pinv(W2) [128, 64]
    __bf16* M6  = (__bf16*)take((size_t)896 * 128 * 2);  // pinv(W1) [784->896, 128]
    __bf16* h1  = (__bf16*)take((size_t)B * 128 * 2);
    __bf16* h2  = (__bf16*)take((size_t)B * 64 * 2);
    __bf16* h3  = (__bf16*)take((size_t)B * 32 * 2);     // [B,16] stored K-padded
    __bf16* h4  = (__bf16*)take((size_t)B * 64 * 2);
    __bf16* h5  = (__bf16*)take((size_t)B * 128 * 2);

    // ---- pseudo-inverse prep: pinv(W) = W^T (W W^T)^-1 ----
    gram_kernel<<<cdiv(128 * 128, 256), 256, 0, stream>>>(W1, G1, 128, 784);
    gram_kernel<<<cdiv(64 * 64, 256), 256, 0, stream>>>(W2, G2, 64, 128);
    gram_kernel<<<1, 256, 0, stream>>>(W3, G3, 16, 64);
    ginv_kernel<<<1, 256, (2 * 128 * 128 + 128) * sizeof(float), stream>>>(G1, G1i, 128);
    ginv_kernel<<<1, 256, (2 * 64 * 64 + 64) * sizeof(float), stream>>>(G2, G2i, 64);
    ginv_kernel<<<1, 256, (2 * 16 * 16 + 16) * sizeof(float), stream>>>(G3, G3i, 16);
    cvt_bf_kernel<<<cdiv(128 * 800, 256), 256, 0, stream>>>(W1, M1, 128, 784, 800, 128 * 800);
    cvt_bf_kernel<<<cdiv(128 * 128, 256), 256, 0, stream>>>(W2, M2, 64, 128, 128, 128 * 128);
    cvt_bf_kernel<<<cdiv(128 * 64, 256), 256, 0, stream>>>(W3, M3, 16, 64, 64, 128 * 64);
    pinv_bf_kernel<<<cdiv(128 * 32, 256), 256, 0, stream>>>(W3, G3i, M4, 16, 64, 32, 128 * 32);
    pinv_bf_kernel<<<cdiv(128 * 64, 256), 256, 0, stream>>>(W2, G2i, M5, 64, 128, 64, 128 * 64);
    pinv_bf_kernel<<<cdiv(896 * 128, 256), 256, 0, stream>>>(W1, G1i, M6, 128, 784, 128, 896 * 128);

    // ---- fused GEMM chain ----
    const int gx = cdiv(B, 64);
    dim3 blk(256);
    // L1: lrelu(x @ W1^T)           [B,784]f32 -> [B,128]bf16
    mlp_gemm_kernel<true, false, true><<<dim3(gx, 1), blk, 0, stream>>>(
        x, M1, h1, B, 784, 784, 800, 128);
    // L2: lrelu(h1 @ W2^T)          [B,128] -> [B,64]
    mlp_gemm_kernel<false, false, true><<<dim3(gx, 1), blk, 0, stream>>>(
        h1, M2, h2, B, 128, 128, 128, 64);
    // L3: h2 @ W3^T                 [B,64] -> [B,16] (stored padded to 32)
    mlp_gemm_kernel<false, false, false><<<dim3(gx, 1), blk, 0, stream>>>(
        h2, M3, h3, B, 64, 64, 64, 32);
    // L4: lrelu(h3 @ d3^T)          [B,16(pad32)] -> [B,64]
    mlp_gemm_kernel<false, false, true><<<dim3(gx, 1), blk, 0, stream>>>(
        h3, M4, h4, B, 32, 32, 32, 64);
    // L5: lrelu(h4 @ d2^T)          [B,64] -> [B,128]
    mlp_gemm_kernel<false, false, true><<<dim3(gx, 1), blk, 0, stream>>>(
        h4, M5, h5, B, 64, 64, 64, 128);
    // L6: h5 @ d1^T                 [B,128] -> [B,784]f32 (output)
    mlp_gemm_kernel<false, true, false><<<dim3(gx, 7), blk, 0, stream>>>(
        h5, M6, d_out, B, 128, 128, 128, 784);
}
